// ProteinClassifier_46755013984572
// MI455X (gfx1250) — compile-verified, matrix operand order checked by reading, not run on hardware
//
#include <hip/hip_runtime.h>
#include <stdint.h>

// ---------------------------------------------------------------------------
// ProteinClassifier for MI455X (gfx1250, wave32).
// GEMM stages (GAT feature transforms, fc1) use v_wmma_f32_16x16x32_bf16 with
// pre-packed bf16 operands: A[M,Kpad] row-major, B transposed to [N,Kpad], so
// each lane's WMMA fragment is exactly two contiguous 16-byte loads
// (global_load_b128) per matrix per K-step -- no divergence, no scalar loads.
// Each wave computes a 16x64 output strip (A fragment reused over 4 WMMAs).
// Irregular segment softmax / scatter ops use VALU + L2 atomics.
//
// Input pointer layout assumption (no runtime to verify):
//   d_in[0] sequence [B,L,F] f32      d_in[1] x_graph [N,F] f32
//   d_in[2] structural [B,62] f32
//   d_in[3..56] params leaves, jax tree_flatten order (sorted dict keys):
//     blocks[0..3] x {bn_b,bn_g,bn_m,bn_v,cb,cw,se1,se2}   -> 3+blk*8 + k
//     bn1 {b,g,m,v} -> 35..38 ; conv1_b 39 ; conv1_w 40
//     fc1_b 41 ; fc1_w 42 ; fc2_b 43 ; fc2_w 44
//     gat[0] {W,a_dst,a_src,b} -> 45..48 ; gat[1] -> 49..52
//     gate_b 53 ; gate_w 54 ; str_b 55 ; str_w 56
//   d_in[57] edge_index [2,E] i32     d_in[58] batch_index [N] i32
// ---------------------------------------------------------------------------

#define B_   64
#define L_   1024
#define F_   21
#define H_   4
#define DH_  64
#define HD_  (H_ * DH_)        // 256
#define GR_  32
#define NB_  4
#define C0_  32
#define CF_  (C0_ + NB_ * GR_) // 160
#define CAT_ (CF_ + HD_ + 32)  // 448
#define EPS_ 1e-5f

typedef __attribute__((ext_vector_type(16))) __bf16 v16bf;
typedef __attribute__((ext_vector_type(8)))  __bf16 v8bf;
typedef __attribute__((ext_vector_type(8)))  float  v8f;

__device__ __forceinline__ __bf16 f2bf(float f) {
  union { float f; uint32_t u; } a; a.f = f;
  uint32_t u = a.u;
  u += 0x7fffu + ((u >> 16) & 1u);       // round-to-nearest-even
  union { uint16_t s; __bf16 h; } b; b.s = (uint16_t)(u >> 16);
  return b.h;
}

// Monotone float<->uint encoding for atomicMax-based segment max.
__device__ __forceinline__ uint32_t fenc(float f) {
  union { float f; uint32_t u; } a; a.f = f;
  return (a.u & 0x80000000u) ? ~a.u : (a.u | 0x80000000u);
}
__device__ __forceinline__ float fdec(uint32_t u) {
  union { uint32_t u; float f; } a;
  a.u = (u & 0x80000000u) ? (u & 0x7fffffffu) : ~u;
  return a.f;
}

// ---------------------------------------------------------------------------
// Operand packing: f32 -> bf16, zero-padded K, B transposed to [N,Kpad].
// ---------------------------------------------------------------------------
__global__ void k_pack_a(const float* __restrict__ in, __bf16* __restrict__ out,
                         int M, int K, int Kpad) {
  int idx = blockIdx.x * blockDim.x + threadIdx.x;
  if (idx >= M * Kpad) return;
  int k = idx % Kpad, m = idx / Kpad;
  out[idx] = f2bf(k < K ? in[(size_t)m * K + k] : 0.0f);
}

__global__ void k_pack_bT(const float* __restrict__ in /*[K,N]*/,
                          __bf16* __restrict__ out /*[N,Kpad]*/,
                          int K, int Nn, int Kpad) {
  int idx = blockIdx.x * blockDim.x + threadIdx.x;
  if (idx >= Nn * Kpad) return;
  int k = idx % Kpad, n = idx / Kpad;
  out[idx] = f2bf(k < K ? in[(size_t)k * Nn + n] : 0.0f);
}

// ---------------------------------------------------------------------------
// WMMA GEMM: C[M,N] = act(A @ B + bias).  A: bf16 [M,Kpad]; Bt: bf16 [N,Kpad].
// 4 waves/block; each wave owns a 16x64 strip (4 accumulators).  Kpad % 32 ==
// 0, M % 16 == 0, N % 64 == 0 (20000/64/256/448 all satisfy).  Per CDNA5 ISA
// 7.12.2, lane (hi=lane>>4, rc=lane&15) needs K runs [kbase, kbase+8) and
// [16+kbase, 16+kbase+8): two contiguous b128 loads per fragment.
// ---------------------------------------------------------------------------
__global__ __launch_bounds__(128)
void k_gemm_wmma(const __bf16* __restrict__ A, const __bf16* __restrict__ Bt,
                 const float* __restrict__ bias, float* __restrict__ C,
                 int Kpad, int Nn, int relu, int ntn /* N/64 */, int ntiles) {
  const int tile = blockIdx.x * 4 + (threadIdx.x >> 5);
  if (tile >= ntiles) return;               // whole-wave uniform exit
  const int lane = threadIdx.x & 31;
  const int hi = lane >> 4, rc = lane & 15;
  const int tm = tile / ntn, tn = tile % ntn;

  const __bf16* ap  = A  + (size_t)(tm * 16 + rc) * Kpad + hi * 8;
  const __bf16* bp0 = Bt + (size_t)(tn * 64 + rc) * Kpad + hi * 8;
  const size_t bstride = (size_t)16 * Kpad;  // next 16-column tile

  v8f acc0 = {}, acc1 = {}, acc2 = {}, acc3 = {};
  for (int k0 = 0; k0 < Kpad; k0 += 32) {
    v8bf al = *(const v8bf*)(ap + k0);
    v8bf ah = *(const v8bf*)(ap + k0 + 16);
    v16bf af = __builtin_shufflevector(al, ah, 0, 1, 2, 3, 4, 5, 6, 7,
                                       8, 9, 10, 11, 12, 13, 14, 15);
    if (k0 + 32 < Kpad)   // next K-slab of A toward L0/L2 (global_prefetch_b8)
      __builtin_prefetch(ap + k0 + 32, 0, 1);

    v8bf bl, bh; v16bf bf;
    bl = *(const v8bf*)(bp0 + k0);
    bh = *(const v8bf*)(bp0 + k0 + 16);
    bf = __builtin_shufflevector(bl, bh, 0,1,2,3,4,5,6,7,8,9,10,11,12,13,14,15);
    acc0 = __builtin_amdgcn_wmma_f32_16x16x32_bf16(false, af, false, bf,
                                                   (short)0, acc0, false, false);
    bl = *(const v8bf*)(bp0 + bstride + k0);
    bh = *(const v8bf*)(bp0 + bstride + k0 + 16);
    bf = __builtin_shufflevector(bl, bh, 0,1,2,3,4,5,6,7,8,9,10,11,12,13,14,15);
    acc1 = __builtin_amdgcn_wmma_f32_16x16x32_bf16(false, af, false, bf,
                                                   (short)0, acc1, false, false);
    bl = *(const v8bf*)(bp0 + 2 * bstride + k0);
    bh = *(const v8bf*)(bp0 + 2 * bstride + k0 + 16);
    bf = __builtin_shufflevector(bl, bh, 0,1,2,3,4,5,6,7,8,9,10,11,12,13,14,15);
    acc2 = __builtin_amdgcn_wmma_f32_16x16x32_bf16(false, af, false, bf,
                                                   (short)0, acc2, false, false);
    bl = *(const v8bf*)(bp0 + 3 * bstride + k0);
    bh = *(const v8bf*)(bp0 + 3 * bstride + k0 + 16);
    bf = __builtin_shufflevector(bl, bh, 0,1,2,3,4,5,6,7,8,9,10,11,12,13,14,15);
    acc3 = __builtin_amdgcn_wmma_f32_16x16x32_bf16(false, af, false, bf,
                                                   (short)0, acc3, false, false);
  }

  v8f accs[4] = {acc0, acc1, acc2, acc3};
#pragma unroll
  for (int j = 0; j < 4; ++j) {
    const int col = tn * 64 + j * 16 + rc;
#pragma unroll
    for (int r = 0; r < 8; ++r) {
      float v = accs[j][r];
      if (bias) v += bias[col];
      if (relu) v = fmaxf(v, 0.0f);
      C[(size_t)(tm * 16 + r + 8 * hi) * Nn + col] = v;
    }
  }
}

// ------------------------- CNN / dense path --------------------------------
__global__ void k_conv1(const float* __restrict__ seq,   // [B,L,F]
                        const float* __restrict__ w,     // [32,F,3]
                        const float* __restrict__ cb,
                        const float* __restrict__ g, const float* __restrict__ bb,
                        const float* __restrict__ m, const float* __restrict__ v,
                        float* __restrict__ concat) {    // [B,CF,L]
  int idx = blockIdx.x * blockDim.x + threadIdx.x;
  if (idx >= B_ * C0_ * L_) return;
  int l  = idx % L_;
  int oc = (idx / L_) % C0_;
  int b  = idx / (L_ * C0_);
  float acc = cb[oc];
  for (int f = 0; f < F_; ++f) {
    const float* wp = &w[(oc * F_ + f) * 3];
    const float* sp = &seq[((size_t)b * L_) * F_ + f];
#pragma unroll
    for (int k = 0; k < 3; ++k) {
      int lp = l + k - 1;
      if (lp >= 0 && lp < L_) acc += wp[k] * sp[(size_t)lp * F_];
    }
  }
  float sc = g[oc] * rsqrtf(v[oc] + EPS_);
  float o  = acc * sc + (bb[oc] - m[oc] * sc);
  concat[((size_t)b * CF_ + oc) * L_ + l] = fmaxf(o, 0.0f);
}

__global__ void k_block_conv(const float* __restrict__ concat,
                             const float* __restrict__ cw,  // [GR,c,3]
                             const float* __restrict__ cb,
                             const float* __restrict__ g, const float* __restrict__ bb,
                             const float* __restrict__ m, const float* __restrict__ v,
                             float* __restrict__ hraw, int c) { // [B,GR,L]
  int idx = blockIdx.x * blockDim.x + threadIdx.x;
  if (idx >= B_ * GR_ * L_) return;
  int l  = idx % L_;
  int oc = (idx / L_) % GR_;
  int b  = idx / (L_ * GR_);
  float acc = cb[oc];
  for (int ci = 0; ci < c; ++ci) {
    float sc = g[ci] * rsqrtf(v[ci] + EPS_);
    float sh = bb[ci] - m[ci] * sc;
    const float* wp = &cw[((size_t)oc * c + ci) * 3];
    const float* ip = &concat[((size_t)b * CF_ + ci) * L_];
#pragma unroll
    for (int k = 0; k < 3; ++k) {
      int lp = l + k - 1;
      if (lp >= 0 && lp < L_) {
        float xv = fmaxf(ip[lp] * sc + sh, 0.0f);   // fused BN+ReLU on read
        acc += wp[k] * xv;
      }
    }
  }
  hraw[((size_t)b * GR_ + oc) * L_ + l] = acc;
}

__global__ __launch_bounds__(256)
void k_mean_over_L(const float* __restrict__ x, float* __restrict__ y) {
  __shared__ float sm[256];
  const float* p = &x[(size_t)blockIdx.x * L_];
  float s = 0.0f;
  for (int l = threadIdx.x; l < L_; l += 256) s += p[l];
  sm[threadIdx.x] = s; __syncthreads();
  for (int st = 128; st > 0; st >>= 1) {
    if (threadIdx.x < st) sm[threadIdx.x] += sm[threadIdx.x + st];
    __syncthreads();
  }
  if (threadIdx.x == 0) y[blockIdx.x] = sm[0] * (1.0f / L_);
}

__global__ __launch_bounds__(32)
void k_se_fc(const float* __restrict__ ymean, const float* __restrict__ se1,
             const float* __restrict__ se2, float* __restrict__ sbuf) {
  // se1:[GR,2]  se2:[2,GR]
  int b = blockIdx.x, t = threadIdx.x;
  __shared__ float y[GR_], tt[2];
  y[t] = ymean[b * GR_ + t];
  __syncthreads();
  if (t < 2) {
    float s = 0.0f;
    for (int i = 0; i < GR_; ++i) s += y[i] * se1[i * 2 + t];
    tt[t] = fmaxf(s, 0.0f);
  }
  __syncthreads();
  float s2 = tt[0] * se2[0 * GR_ + t] + tt[1] * se2[1 * GR_ + t];
  sbuf[b * GR_ + t] = 1.0f / (1.0f + expf(-s2));
}

__global__ void k_se_scale(const float* __restrict__ hraw,
                           const float* __restrict__ sbuf,
                           float* __restrict__ concat, int coff) {
  int idx = blockIdx.x * blockDim.x + threadIdx.x;
  if (idx >= B_ * GR_ * L_) return;
  int l  = idx % L_;
  int gc = (idx / L_) % GR_;
  int b  = idx / (L_ * GR_);
  concat[((size_t)b * CF_ + coff + gc) * L_ + l] =
      hraw[((size_t)b * GR_ + gc) * L_ + l] * sbuf[b * GR_ + gc];
}

// ------------------------------ GAT path -----------------------------------
__global__ void k_attn_sd(const float* __restrict__ hfeat,
                          const float* __restrict__ a_src,
                          const float* __restrict__ a_dst,
                          float* __restrict__ sv, float* __restrict__ dv, int N) {
  int idx = blockIdx.x * blockDim.x + threadIdx.x;
  if (idx >= N * H_) return;
  int h = idx % H_, n = idx / H_;
  const float* hp = &hfeat[(size_t)n * HD_ + h * DH_];
  float s = 0.0f, d = 0.0f;
  for (int k = 0; k < DH_; ++k) {
    float x = hp[k];
    s += x * a_src[h * DH_ + k];
    d += x * a_dst[h * DH_ + k];
  }
  sv[idx] = s; dv[idx] = d;
}

__global__ void k_edge_att(const int* __restrict__ ei, int E, int Etot,
                           const float* __restrict__ sv, const float* __restrict__ dv,
                           float* __restrict__ ebuf, uint32_t* __restrict__ menc) {
  int idx = blockIdx.x * blockDim.x + threadIdx.x;
  if (idx >= Etot * H_) return;
  int h = idx % H_, e = idx / H_;
  int src = (e < E) ? ei[e]     : (e - E);
  int dst = (e < E) ? ei[E + e] : (e - E);
  float x = sv[src * H_ + h] + dv[dst * H_ + h];
  x = (x > 0.0f) ? x : 0.2f * x;          // leaky_relu 0.2
  ebuf[idx] = x;
  atomicMax(&menc[dst * H_ + h], fenc(x));
}

__global__ void k_edge_exp(const int* __restrict__ ei, int E, int Etot,
                           const float* __restrict__ ebuf,
                           const uint32_t* __restrict__ menc,
                           float* __restrict__ exbuf, float* __restrict__ den) {
  int idx = blockIdx.x * blockDim.x + threadIdx.x;
  if (idx >= Etot * H_) return;
  int h = idx % H_, e = idx / H_;
  int dst = (e < E) ? ei[E + e] : (e - E);
  float m = fdec(menc[dst * H_ + h]);
  if (!(m == m) || fabsf(m) > 3.0e38f) m = 0.0f;  // mirror isfinite fixup
  float ex = expf(ebuf[idx] - m);
  exbuf[idx] = ex;
  atomicAdd(&den[dst * H_ + h], ex);
}

__global__ void k_edge_agg(const int* __restrict__ ei, int E, int Etot,
                           const float* __restrict__ exbuf,
                           const float* __restrict__ den,
                           const float* __restrict__ hfeat,
                           float* __restrict__ agg) {
  int idx = blockIdx.x * blockDim.x + threadIdx.x;
  if (idx >= Etot * H_) return;
  int h = idx % H_, e = idx / H_;
  int src = (e < E) ? ei[e]     : (e - E);
  int dst = (e < E) ? ei[E + e] : (e - E);
  float w = exbuf[idx] / (den[dst * H_ + h] + 1e-16f);
  const float* hp = &hfeat[(size_t)src * HD_ + h * DH_];
  float* op = &agg[(size_t)dst * HD_ + h * DH_];
  for (int k = 0; k < DH_; ++k) atomicAdd(&op[k], w * hp[k]);
}

__global__ void k_bias_relu(const float* __restrict__ agg,
                            const float* __restrict__ bias,
                            float* __restrict__ out, int total) {
  int idx = blockIdx.x * blockDim.x + threadIdx.x;
  if (idx >= total) return;
  out[idx] = fmaxf(agg[idx] + bias[idx % HD_], 0.0f);
}

// ------------------------- attention pooling -------------------------------
__global__ void k_gate(const float* __restrict__ xg, const float* __restrict__ gw,
                       const float* __restrict__ gb, float* __restrict__ gate, int N) {
  int n = blockIdx.x * blockDim.x + threadIdx.x;
  if (n >= N) return;
  float s = gb[0];
  const float* p = &xg[(size_t)n * HD_];
  for (int j = 0; j < HD_; ++j) s += p[j] * gw[j];
  gate[n] = s;
}

__global__ void k_gate_max(const float* __restrict__ gate, const int* __restrict__ bi,
                           uint32_t* __restrict__ gmenc, int N) {
  int n = blockIdx.x * blockDim.x + threadIdx.x;
  if (n >= N) return;
  atomicMax(&gmenc[bi[n]], fenc(gate[n]));
}

__global__ void k_gate_exp(const float* __restrict__ gate, const int* __restrict__ bi,
                           const uint32_t* __restrict__ gmenc,
                           float* __restrict__ ge, float* __restrict__ gs, int N) {
  int n = blockIdx.x * blockDim.x + threadIdx.x;
  if (n >= N) return;
  float v = expf(gate[n] - fdec(gmenc[bi[n]]));
  ge[n] = v;
  atomicAdd(&gs[bi[n]], v);
}

__global__ void k_gate_pool(const float* __restrict__ ge, const float* __restrict__ gs,
                            const int* __restrict__ bi, const float* __restrict__ xg,
                            float* __restrict__ pooled, int N) {
  int n = blockIdx.x * blockDim.x + threadIdx.x;
  if (n >= N) return;
  int b = bi[n];
  float w = ge[n] / (gs[b] + 1e-16f);
  const float* p = &xg[(size_t)n * HD_];
  float* o = &pooled[(size_t)b * HD_];
  for (int j = 0; j < HD_; ++j) atomicAdd(&o[j], w * p[j]);
}

// ----------------------------- tail ----------------------------------------
__global__ void k_struct(const float* __restrict__ sf, const float* __restrict__ w,
                         const float* __restrict__ b, float* __restrict__ xs) {
  int idx = blockIdx.x * blockDim.x + threadIdx.x;
  if (idx >= B_ * 32) return;
  int j = idx % 32, bb = idx / 32;
  float s = b[j];
  for (int i = 0; i < 62; ++i) s += sf[bb * 62 + i] * w[i * 32 + j];
  xs[idx] = fmaxf(s, 0.0f);
}

__global__ void k_cat(const float* __restrict__ outp, const float* __restrict__ pooled,
                      const float* __restrict__ xs, float* __restrict__ cat) {
  int idx = blockIdx.x * blockDim.x + threadIdx.x;
  if (idx >= B_ * CAT_) return;
  int j = idx % CAT_, b = idx / CAT_;
  float v;
  if (j < CF_)            v = outp[b * CF_ + j];
  else if (j < CF_ + HD_) v = pooled[b * HD_ + (j - CF_)];
  else                    v = xs[b * 32 + (j - CF_ - HD_)];
  cat[idx] = v;
}

__global__ __launch_bounds__(256)
void k_fc2(const float* __restrict__ h, const float* __restrict__ w,
           const float* __restrict__ b, float* __restrict__ out) {
  __shared__ float sm[256];
  int bb = blockIdx.x, t = threadIdx.x;
  sm[t] = h[(size_t)bb * HD_ + t] * w[t];
  __syncthreads();
  for (int st = 128; st > 0; st >>= 1) {
    if (t < st) sm[t] += sm[t + st];
    __syncthreads();
  }
  if (t == 0) out[bb] = 1.0f / (1.0f + expf(-(sm[0] + b[0])));
}

// ---------------------------------------------------------------------------
extern "C" void kernel_launch(void* const* d_in, const int* in_sizes, int n_in,
                              void* d_out, int out_size, void* d_ws, size_t ws_size,
                              hipStream_t stream) {
  (void)n_in; (void)out_size; (void)ws_size;
  const float* seq = (const float*)d_in[0];
  const float* xgr = (const float*)d_in[1];
  const float* sf  = (const float*)d_in[2];
  const int*   ei  = (const int*)d_in[57];
  const int*   bi  = (const int*)d_in[58];
  const int N    = in_sizes[1] / F_;    // 20000
  const int E    = in_sizes[57] / 2;    // 320000
  const int Etot = E + N;

#define P(i) ((const float*)d_in[(i)])

  char* ws = (char*)d_ws;
  size_t off = 0;
  auto alloc = [&](size_t bytes) {
    char* p = ws + off;
    off += (bytes + 255) & ~(size_t)255;
    return p;
  };
  float*    concat = (float*)alloc((size_t)B_ * CF_ * L_ * 4);
  float*    hraw   = (float*)alloc((size_t)B_ * GR_ * L_ * 4);
  float*    ymean  = (float*)alloc((size_t)B_ * GR_ * 4);
  float*    sbuf   = (float*)alloc((size_t)B_ * GR_ * 4);
  float*    outp   = (float*)alloc((size_t)B_ * CF_ * 4);
  float*    hfeat  = (float*)alloc((size_t)N * HD_ * 4);
  float*    xga    = (float*)alloc((size_t)N * HD_ * 4);
  float*    agg    = (float*)alloc((size_t)N * HD_ * 4);
  __bf16*   abf    = (__bf16*)alloc((size_t)N * HD_ * 2);   // packed A (bf16)
  __bf16*   btbf   = (__bf16*)alloc((size_t)HD_ * CAT_ * 2); // packed B^T (bf16)
  float*    sv     = (float*)alloc((size_t)N * H_ * 4);
  float*    dv     = (float*)alloc((size_t)N * H_ * 4);
  float*    ebuf   = (float*)alloc((size_t)Etot * H_ * 4);
  float*    exbuf  = (float*)alloc((size_t)Etot * H_ * 4);
  uint32_t* menc   = (uint32_t*)alloc((size_t)N * H_ * 4);
  float*    den    = (float*)alloc((size_t)N * H_ * 4);
  float*    gate   = (float*)alloc((size_t)N * 4);
  float*    ge     = (float*)alloc((size_t)N * 4);
  uint32_t* gmenc  = (uint32_t*)alloc((size_t)B_ * 4);
  float*    gs     = (float*)alloc((size_t)B_ * 4);
  float*    pooled = (float*)alloc((size_t)B_ * HD_ * 4);
  float*    xs     = (float*)alloc((size_t)B_ * 32 * 4);
  float*    cat    = (float*)alloc((size_t)B_ * CAT_ * 4);
  float*    fc1h   = (float*)alloc((size_t)B_ * HD_ * 4);

  const int T = 256;
  auto nb = [](long long n, int t) { return (unsigned)((n + t - 1) / t); };
  // packed-bf16 WMMA GEMM launcher: C[M,Nn] = act(A[M,K] @ Bw[K,Nn] + bias)
  auto gemm = [&](const float* A, int M, int K, int Kpad, const float* Bw,
                  int Nn, const float* bias, float* C, int relu) {
    k_pack_a<<<nb((long long)M * Kpad, T), T, 0, stream>>>(A, abf, M, K, Kpad);
    k_pack_bT<<<nb((long long)Nn * Kpad, T), T, 0, stream>>>(Bw, btbf, K, Nn, Kpad);
    const int ntn = Nn / 64;
    const int ntiles = (M / 16) * ntn;
    k_gemm_wmma<<<nb(ntiles, 4), 128, 0, stream>>>(abf, btbf, bias, C,
                                                   Kpad, Nn, relu, ntn, ntiles);
  };

  // ---- CNN path ----
  k_conv1<<<nb((long long)B_ * C0_ * L_, T), T, 0, stream>>>(
      seq, P(40), P(39), P(36), P(35), P(37), P(38), concat);
  int c = C0_;
  for (int blk = 0; blk < NB_; ++blk) {
    const int base = 3 + blk * 8;
    const int coff = C0_ + blk * GR_;
    k_block_conv<<<nb((long long)B_ * GR_ * L_, T), T, 0, stream>>>(
        concat, P(base + 5), P(base + 4), P(base + 1), P(base + 0),
        P(base + 2), P(base + 3), hraw, c);
    k_mean_over_L<<<B_ * GR_, 256, 0, stream>>>(hraw, ymean);
    k_se_fc<<<B_, 32, 0, stream>>>(ymean, P(base + 6), P(base + 7), sbuf);
    k_se_scale<<<nb((long long)B_ * GR_ * L_, T), T, 0, stream>>>(hraw, sbuf, concat, coff);
    c += GR_;
  }
  k_mean_over_L<<<B_ * CF_, 256, 0, stream>>>(concat, outp);

  // ---- GAT layers (bf16 WMMA feature transform + atomic segment softmax) ----
  for (int layer = 0; layer < 2; ++layer) {
    const int gbase = 45 + layer * 4;              // W, a_dst, a_src, b
    if (layer == 0) gemm(xgr, N, F_, 32, P(gbase + 0), HD_, nullptr, hfeat, 0);
    else            gemm(xga, N, HD_, HD_, P(gbase + 0), HD_, nullptr, hfeat, 0);
    k_attn_sd<<<nb((long long)N * H_, T), T, 0, stream>>>(
        hfeat, P(gbase + 2), P(gbase + 1), sv, dv, N);
    hipMemsetAsync(menc, 0, (size_t)N * H_ * 4, stream);
    hipMemsetAsync(den,  0, (size_t)N * H_ * 4, stream);
    hipMemsetAsync(agg,  0, (size_t)N * HD_ * 4, stream);
    k_edge_att<<<nb((long long)Etot * H_, T), T, 0, stream>>>(ei, E, Etot, sv, dv, ebuf, menc);
    k_edge_exp<<<nb((long long)Etot * H_, T), T, 0, stream>>>(ei, E, Etot, ebuf, menc, exbuf, den);
    k_edge_agg<<<nb((long long)Etot * H_, T), T, 0, stream>>>(ei, E, Etot, exbuf, den, hfeat, agg);
    k_bias_relu<<<nb((long long)N * HD_, T), T, 0, stream>>>(agg, P(gbase + 3), xga, N * HD_);
  }

  // ---- global attention pooling ----
  k_gate<<<nb(N, T), T, 0, stream>>>(xga, P(54), P(53), gate, N);
  hipMemsetAsync(gmenc, 0, (size_t)B_ * 4, stream);
  hipMemsetAsync(gs,    0, (size_t)B_ * 4, stream);
  hipMemsetAsync(pooled, 0, (size_t)B_ * HD_ * 4, stream);
  k_gate_max<<<nb(N, T), T, 0, stream>>>(gate, bi, gmenc, N);
  k_gate_exp<<<nb(N, T), T, 0, stream>>>(gate, bi, gmenc, ge, gs, N);
  k_gate_pool<<<nb(N, T), T, 0, stream>>>(ge, gs, bi, xga, pooled, N);

  // ---- tail: struct MLP, concat, fc1 (WMMA, bias+relu fused), fc2+sigmoid ----
  k_struct<<<nb((long long)B_ * 32, T), T, 0, stream>>>(sf, P(56), P(55), xs);
  k_cat<<<nb((long long)B_ * CAT_, T), T, 0, stream>>>(outp, pooled, xs, cat);
  gemm(cat, B_, CAT_, CAT_, P(42), HD_, P(41), fc1h, 1);
  k_fc2<<<B_, 256, 0, stream>>>(fc1h, P(44), P(43), (float*)d_out);
#undef P
}
